// MultiHeadAttention_9715216024245
// MI455X (gfx1250) — matrix operand order repaired and checked by reading, not compiled, not run on hardware
//
#include <hip/hip_runtime.h>
#include <hip/hip_bf16.h>

// ---- problem constants (match reference) ----
#define SEQ_LEN   2048
#define NUM_HEADS 16
#define DIM_EMB   1024
#define DIM_HEAD  64
#define BATCH     2
#define ROWS      (BATCH * SEQ_LEN)   // 4096 total sequence rows
#define MB        4                   // M-subtiles (rows of 16) per wave

typedef __attribute__((ext_vector_type(16))) __bf16 v16bf;
typedef __attribute__((ext_vector_type(8)))  __bf16 v8bf;
typedef __attribute__((ext_vector_type(8)))  float  v8f;
typedef unsigned int u32x4 __attribute__((ext_vector_type(4)));
typedef int          i32x4 __attribute__((ext_vector_type(4)));
typedef int          i32x8 __attribute__((ext_vector_type(8)));

#ifndef __has_builtin
#define __has_builtin(x) 0
#endif
#if __has_builtin(__builtin_amdgcn_tensor_load_to_lds) && \
    __has_builtin(__builtin_amdgcn_s_wait_tensorcnt)
#define USE_TDM 1
#else
#define USE_TDM 0
#endif

// D = A(16x32 bf16) * B(32x16 bf16) + C(16x16 f32)
static __device__ __forceinline__ v8f wmma_bf16(v16bf a, v16bf b, v8f c) {
  return __builtin_amdgcn_wmma_f32_16x16x32_bf16(false, a, false, b, (short)0, c, false, false);
}

// A-fragment (16x32, 16-bit): lane L, element h:
//   K = 16*(L/16) + h,  M = L%16
// p must already point at  row(M)*ld + k0 + 16*(L>>4)   (16B aligned)
static __device__ __forceinline__ v16bf load_afrag(const __bf16* p) {
  v8bf lo = *reinterpret_cast<const v8bf*>(p);
  v8bf hi = *reinterpret_cast<const v8bf*>(p + 8);
  v16bf a;
#pragma unroll
  for (int i = 0; i < 8; ++i) { a[i] = lo[i]; a[i + 8] = hi[i]; }
  return a;
}

#if USE_TDM
// Generic pointer to LDS byte offset: ISA 10.2, LDS aperture addr[31:0] is the
// LDS address, so truncation gives the workgroup-relative LDS byte offset.
static __device__ __forceinline__ unsigned lds_off(const void* p) {
  return (unsigned)(unsigned long long)p;
}

// Issue a 2-D TDM tile load (bf16 elements): copies tile_d1 rows of tile_d0
// contiguous elements (row stride d0_stride) from gaddr into LDS at lds_addr.
// D# packing per CDNA5 ISA.
static __device__ __forceinline__ void tdm_load_2d(
    unsigned lds_addr, const void* gaddr,
    unsigned tensor_d0, unsigned tensor_d1,
    unsigned tile_d0, unsigned tile_d1, unsigned d0_stride)
{
  const unsigned long long ga = (unsigned long long)gaddr;
  u32x4 g0;
  g0[0] = 1u;                                         // count=1, user mode
  g0[1] = lds_addr;                                   // lds_addr [63:32]
  g0[2] = (unsigned)(ga & 0xffffffffu);               // global_addr lo
  g0[3] = (unsigned)((ga >> 32) & 0x01ffffffu)        // global_addr [56:32]
          | 0x80000000u;                              // type=2 ("image")
  i32x8 g1;
  g1[0] = (int)(1u << 16);                            // data_size=1 (2 bytes)
  g1[1] = (int)((tensor_d0 & 0xffffu) << 16);         // tensor_dim0 [63:48]
  g1[2] = (int)(((tensor_d0 >> 16) & 0xffffu) |       // tensor_dim0 [79:64]
                ((tensor_d1 & 0xffffu) << 16));       // tensor_dim1 [95:80]
  g1[3] = (int)(((tensor_d1 >> 16) & 0xffffu) |       // tensor_dim1 [111:96]
                ((tile_d0 & 0xffffu) << 16));         // tile_dim0  [127:112]
  g1[4] = (int)(tile_d1 & 0xffffu);                   // tile_dim1; tile_dim2=0
  g1[5] = (int)d0_stride;                             // tensor_dim0_stride lo
  g1[6] = 0;                                          // stride hi / dim1_stride
  g1[7] = 0;
  const i32x4 z4 = {0, 0, 0, 0};
#if __clang_major__ >= 23
  const i32x8 z8 = {0, 0, 0, 0, 0, 0, 0, 0};
  __builtin_amdgcn_tensor_load_to_lds(g0, g1, z4, z4, z8, 0);
#else
  __builtin_amdgcn_tensor_load_to_lds(g0, g1, z4, z4, 0);
#endif
}
#endif // USE_TDM

// ------------------------------------------------------------------
// prep: fp32 -> bf16 straight copy
__global__ void cvt_f32_to_bf16(const float* __restrict__ src,
                                __bf16* __restrict__ dst, int n) {
  int i = blockIdx.x * blockDim.x + threadIdx.x;
  int stride = gridDim.x * blockDim.x;
  for (; i < n; i += stride) dst[i] = (__bf16)src[i];
}

// prep: src[K][N] fp32 -> dst[N][K] bf16 (transposed weights => contiguous
// WMMA B-fragment loads)
__global__ void cvt_transpose_bf16(const float* __restrict__ src,
                                   __bf16* __restrict__ dst, int K, int N) {
  long total = (long)K * N;
  long i = (long)blockIdx.x * blockDim.x + threadIdx.x;
  long stride = (long)gridDim.x * blockDim.x;
  for (; i < total; i += stride) {
    int nn = (int)(i / K);
    int kk = (int)(i % K);
    dst[i] = (__bf16)src[(long)kk * N + nn];
  }
}

// ------------------------------------------------------------------
// QKV projection: 64(M) x 64(N) tile per wave (4 M-subtiles reuse each
// B-fragment), fused RoPE + layout shuffles.
// grid: (ROWS/64, 48); nc 0..15 = Q heads, 16..31 = K heads, 32..47 = V heads
__global__ __launch_bounds__(32) void qkv_gemm_rope(
    const __bf16* __restrict__ Xb,     // [ROWS][1024]
    const __bf16* __restrict__ WqkvT,  // [3072][1024]  (transposed)
    __bf16* __restrict__ Qb,           // [B][H][S][64]  (RoPE'd)
    __bf16* __restrict__ Kb,           // [B][H][S][64]  (RoPE'd)
    __bf16* __restrict__ VT)           // [B][H][64][S]  (transposed)
{
  const int lane = threadIdx.x;
  const int lm = lane & 15, lh = lane >> 4;
  const int row0 = blockIdx.x * (16 * MB);
  const int nc = blockIdx.y;
  const int n0 = nc * 64;

  v8f acc[MB][4];
#pragma unroll
  for (int m = 0; m < MB; ++m)
#pragma unroll
    for (int f = 0; f < 4; ++f) acc[m][f] = v8f{};

  for (int k0 = 0; k0 < DIM_EMB; k0 += 32) {
    v16bf a[MB];
#pragma unroll
    for (int m = 0; m < MB; ++m)
      a[m] = load_afrag(Xb + (long)(row0 + m * 16 + lm) * DIM_EMB + k0 + lh * 16);
#pragma unroll
    for (int f = 0; f < 4; ++f) {
      const __bf16* bp = WqkvT + (long)(n0 + f * 16 + lm) * DIM_EMB + k0 + lh * 16;
      const v16bf bb = *reinterpret_cast<const v16bf*>(bp);
#pragma unroll
      for (int m = 0; m < MB; ++m) acc[m][f] = wmma_bf16(a[m], bb, acc[m][f]);
    }
  }

  const int b  = row0 >> 11;             // 64-aligned rows: no batch straddle
  const int s0 = row0 & (SEQ_LEN - 1);

  if (nc < 32) {  // Q or K chunk: RoPE in-register, store [B,H,S,64]
    const int h = nc & 15;
    __bf16* dstbase = (nc < 16 ? Qb : Kb) +
                      (long)(b * NUM_HEADS + h) * SEQ_LEN * DIM_HEAD;
#pragma unroll
    for (int f = 0; f < 4; ++f) {
      const int d = f * 16 + lm;         // 0..63 within head
      const int j = d & 31;
      const float inv = 1.0f / (__powf(10000.0f, (float)j * (1.0f / 32.0f)) +
                                1.1920928955078125e-7f);
#pragma unroll
      for (int m = 0; m < MB; ++m)
#pragma unroll
        for (int r = 0; r < 8; ++r) {
          const int srow = s0 + m * 16 + r + 8 * lh;
          const float ang = (float)srow * inv;
          const float c = __cosf(ang), sn = __sinf(ang);
          const float xv  = acc[m][f][r];
          // rotate_half: first half -> -x[d+32] (frag f+2), second -> x[d-32]
          const float rot = (f < 2) ? -acc[m][f + 2][r] : acc[m][f - 2][r];
          dstbase[(long)srow * DIM_HEAD + d] = (__bf16)(xv * c + rot * sn);
        }
    }
  } else {        // V chunk: store transposed [B,H,64,S]
    const int h = nc - 32;
    __bf16* dstbase = VT + (long)(b * NUM_HEADS + h) * DIM_HEAD * SEQ_LEN;
#pragma unroll
    for (int f = 0; f < 4; ++f) {
      const int d = f * 16 + lm;
#pragma unroll
      for (int m = 0; m < MB; ++m)
#pragma unroll
        for (int r = 0; r < 8; ++r)
          dstbase[(long)d * SEQ_LEN + (s0 + m * 16 + r + 8 * lh)] =
              (__bf16)acc[m][f][r];
    }
  }
}

// ------------------------------------------------------------------
// Causal flash attention: one wave per 64-query tile (4 M-subtiles), online
// softmax; K/V tiles staged into LDS by the Tensor Data Mover, double-buffered
// one tile ahead (s_wait_tensorcnt 2 = overlap DMA with compute).
// grid: (S/64, H, B)
__global__ __launch_bounds__(32) void flash_attn(
    const __bf16* __restrict__ Qb, const __bf16* __restrict__ Kb,
    const __bf16* __restrict__ VT, __bf16* __restrict__ Om /* [ROWS][1024] */)
{
  const int lane = threadIdx.x;
  const int lm = lane & 15, lh = lane >> 4;
  const int q0 = blockIdx.x * (16 * MB);
  const int h = blockIdx.y, b = blockIdx.z;
  const long headoff = (long)(b * NUM_HEADS + h) * SEQ_LEN * DIM_HEAD;
  const __bf16* Qh = Qb + headoff;
  const __bf16* Kh = Kb + headoff;
  const __bf16* Vh = VT + headoff;   // [64][S]

  __shared__ __attribute__((aligned(32))) __bf16 pLds[MB][16][40];
#if USE_TDM
  __shared__ __attribute__((aligned(32))) __bf16 kTile[2][32 * DIM_HEAD];
  __shared__ __attribute__((aligned(32))) __bf16 vTile[2][DIM_HEAD * 32];
#define ISSUE_TILE(j) do {                                                   \
    const int _kk = (j) * 32;                                                \
    tdm_load_2d(lds_off(&kTile[(j) & 1][0]), Kh + (long)_kk * DIM_HEAD,      \
                DIM_HEAD, (unsigned)(SEQ_LEN - _kk), DIM_HEAD, 32, DIM_HEAD);\
    tdm_load_2d(lds_off(&vTile[(j) & 1][0]), Vh + _kk,                       \
                (unsigned)(SEQ_LEN - _kk), DIM_HEAD, 32, DIM_HEAD, SEQ_LEN); \
  } while (0)
#endif

  v16bf qa[MB][2];
#pragma unroll
  for (int m = 0; m < MB; ++m)
#pragma unroll
    for (int t = 0; t < 2; ++t)
      qa[m][t] =
          load_afrag(Qh + (long)(q0 + m * 16 + lm) * DIM_HEAD + t * 32 + lh * 16);

  float mrun[MB][8], lrun[MB][8];
  v8f o[MB][4];
#pragma unroll
  for (int m = 0; m < MB; ++m) {
#pragma unroll
    for (int r = 0; r < 8; ++r) { mrun[m][r] = -1e30f; lrun[m][r] = 0.0f; }
#pragma unroll
    for (int f = 0; f < 4; ++f) o[m][f] = v8f{};
  }
  const float scale = 0.03125f;      // DIM_K^-0.5 = 1/32

  const int ktiles = ((q0 + 16 * MB - 1) >> 5) + 1;   // 32 keys per tile

#if USE_TDM
  ISSUE_TILE(0);
#endif
  for (int i = 0; i < ktiles; ++i) {
    const int kk0 = i * 32;
#if USE_TDM
    if (i + 1 < ktiles) {
      ISSUE_TILE(i + 1);
      __builtin_amdgcn_s_wait_tensorcnt(2);  // current tile's 2 DMAs done
    } else {
      __builtin_amdgcn_s_wait_tensorcnt(0);
    }
    __syncthreads();
    const __bf16* kbase = &kTile[i & 1][0];
    const __bf16* vbase = &vTile[i & 1][0];
#endif
#pragma unroll
    for (int m = 0; m < MB; ++m) {
      float sv[2][8];
#pragma unroll
      for (int sub = 0; sub < 2; ++sub) {
        v8f s = v8f{};
#pragma unroll
        for (int t = 0; t < 2; ++t) {   // head-dim 64 = 2 K-steps of 32
#if USE_TDM
          const v16bf bb = *reinterpret_cast<const v16bf*>(
              kbase + (sub * 16 + lm) * DIM_HEAD + t * 32 + lh * 16);
#else
          const v16bf bb = *reinterpret_cast<const v16bf*>(
              Kh + (long)(kk0 + sub * 16 + lm) * DIM_HEAD + t * 32 + lh * 16);
#endif
          s = wmma_bf16(qa[m][t], bb, s);
        }
        const int col = kk0 + sub * 16 + lm;
#pragma unroll
        for (int r = 0; r < 8; ++r) {
          const int row = q0 + m * 16 + r + 8 * lh;
          sv[sub][r] = (col > row) ? -1e30f : s[r] * scale;
        }
      }
      // online softmax: row reductions across the 16 lanes holding each row
#pragma unroll
      for (int r = 0; r < 8; ++r) {
        float mx = fmaxf(sv[0][r], sv[1][r]);
        mx = fmaxf(mx, __shfl_xor(mx, 1, 32));
        mx = fmaxf(mx, __shfl_xor(mx, 2, 32));
        mx = fmaxf(mx, __shfl_xor(mx, 4, 32));
        mx = fmaxf(mx, __shfl_xor(mx, 8, 32));
        const float mnew = fmaxf(mrun[m][r], mx);
        const float al = __expf(mrun[m][r] - mnew);
        mrun[m][r] = mnew;
        const float e0 = __expf(sv[0][r] - mnew);
        const float e1 = __expf(sv[1][r] - mnew);
        sv[0][r] = e0; sv[1][r] = e1;
        float ts = e0 + e1;
        ts += __shfl_xor(ts, 1, 32);
        ts += __shfl_xor(ts, 2, 32);
        ts += __shfl_xor(ts, 4, 32);
        ts += __shfl_xor(ts, 8, 32);
        lrun[m][r] = lrun[m][r] * al + ts;
#pragma unroll
        for (int f = 0; f < 4; ++f) o[m][f][r] *= al;   // rescale accumulator
      }
      // P: C-layout -> row-major LDS (per-msub tile) for A-fragment reload
#pragma unroll
      for (int sub = 0; sub < 2; ++sub)
#pragma unroll
        for (int r = 0; r < 8; ++r)
          pLds[m][r + 8 * lh][sub * 16 + lm] = (__bf16)sv[sub][r];
    }
    __syncthreads();
    // V B-fragments loaded once, reused by all 4 M-subtiles
    v16bf vb[4];
#pragma unroll
    for (int f = 0; f < 4; ++f) {
#if USE_TDM
      vb[f] = *reinterpret_cast<const v16bf*>(vbase + (f * 16 + lm) * 32 + lh * 16);
#else
      vb[f] = *reinterpret_cast<const v16bf*>(
          Vh + (long)(f * 16 + lm) * SEQ_LEN + kk0 + lh * 16);
#endif
    }
#pragma unroll
    for (int m = 0; m < MB; ++m) {
      const v16bf pa = load_afrag(&pLds[m][lm][0] + lh * 16);
#pragma unroll
      for (int f = 0; f < 4; ++f) o[m][f] = wmma_bf16(pa, vb[f], o[m][f]);
    }
    __syncthreads();
  }
  // normalize and store heads-concatenated bf16 matrix [ROWS][1024]
#pragma unroll
  for (int m = 0; m < MB; ++m)
#pragma unroll
    for (int f = 0; f < 4; ++f)
#pragma unroll
      for (int r = 0; r < 8; ++r) {
        const int row = q0 + m * 16 + r + 8 * lh;
        const int col = h * DIM_HEAD + f * 16 + lm;
        Om[(long)(b * SEQ_LEN + row) * DIM_EMB + col] =
            (__bf16)(o[m][f][r] / lrun[m][r]);
      }
}

// ------------------------------------------------------------------
// out = Om(bf16) @ Wout, fp32 result, 64x64 tile per wave. grid: (ROWS/64, 16)
__global__ __launch_bounds__(32) void out_gemm(
    const __bf16* __restrict__ Om, const __bf16* __restrict__ WoutT,
    float* __restrict__ out)
{
  const int lane = threadIdx.x;
  const int lm = lane & 15, lh = lane >> 4;
  const int row0 = blockIdx.x * (16 * MB);
  const int n0 = blockIdx.y * 64;
  v8f acc[MB][4];
#pragma unroll
  for (int m = 0; m < MB; ++m)
#pragma unroll
    for (int f = 0; f < 4; ++f) acc[m][f] = v8f{};

  for (int k0 = 0; k0 < DIM_EMB; k0 += 32) {
    v16bf a[MB];
#pragma unroll
    for (int m = 0; m < MB; ++m)
      a[m] = load_afrag(Om + (long)(row0 + m * 16 + lm) * DIM_EMB + k0 + lh * 16);
#pragma unroll
    for (int f = 0; f < 4; ++f) {
      const __bf16* bp = WoutT + (long)(n0 + f * 16 + lm) * DIM_EMB + k0 + lh * 16;
      const v16bf bb = *reinterpret_cast<const v16bf*>(bp);
#pragma unroll
      for (int m = 0; m < MB; ++m) acc[m][f] = wmma_bf16(a[m], bb, acc[m][f]);
    }
  }
#pragma unroll
  for (int m = 0; m < MB; ++m)
#pragma unroll
    for (int f = 0; f < 4; ++f)
#pragma unroll
      for (int r = 0; r < 8; ++r)
        out[(long)(row0 + m * 16 + r + 8 * lh) * DIM_EMB + n0 + f * 16 + lm] =
            acc[m][f][r];
}

// ------------------------------------------------------------------
extern "C" void kernel_launch(void* const* d_in, const int* in_sizes, int n_in,
                              void* d_out, int out_size, void* d_ws, size_t ws_size,
                              hipStream_t stream) {
  const float* x    = (const float*)d_in[0];  // [2,2048,1024]
  const float* Wqkv = (const float*)d_in[1];  // [1024,3072]
  const float* Wout = (const float*)d_in[2];  // [1024,1024]
  float* out = (float*)d_out;

  // workspace carve-out (~48 MB total)
  char* ws = (char*)d_ws;
  size_t off = 0;
  auto carve = [&](size_t bytes) -> void* {
    void* p = ws + off;
    off += (bytes + 255) & ~(size_t)255;
    return p;
  };
  __bf16* Xb    = (__bf16*)carve((size_t)ROWS * DIM_EMB * 2);          // 8 MB
  __bf16* WqkvT = (__bf16*)carve((size_t)3 * DIM_EMB * DIM_EMB * 2);   // 6 MB
  __bf16* WoutT = (__bf16*)carve((size_t)DIM_EMB * DIM_EMB * 2);       // 2 MB
  __bf16* Qb    = (__bf16*)carve((size_t)ROWS * DIM_EMB * 2);          // 8 MB
  __bf16* Kb    = (__bf16*)carve((size_t)ROWS * DIM_EMB * 2);          // 8 MB
  __bf16* VT    = (__bf16*)carve((size_t)ROWS * DIM_EMB * 2);          // 8 MB
  __bf16* Om    = (__bf16*)carve((size_t)ROWS * DIM_EMB * 2);          // 8 MB
  (void)ws_size; (void)in_sizes; (void)n_in; (void)out_size;

  cvt_f32_to_bf16<<<4096, 256, 0, stream>>>(x, Xb, ROWS * DIM_EMB);
  cvt_transpose_bf16<<<4096, 256, 0, stream>>>(Wqkv, WqkvT, DIM_EMB, 3 * DIM_EMB);
  cvt_transpose_bf16<<<2048, 256, 0, stream>>>(Wout, WoutT, DIM_EMB, DIM_EMB);
  qkv_gemm_rope<<<dim3(ROWS / (16 * MB), 48), 32, 0, stream>>>(Xb, WqkvT, Qb, Kb, VT);
  flash_attn<<<dim3(SEQ_LEN / (16 * MB), NUM_HEADS, BATCH), 32, 0, stream>>>(Qb, Kb, VT, Om);
  out_gemm<<<dim3(ROWS / (16 * MB), DIM_EMB / 64), 32, 0, stream>>>(Om, WoutT, out);
}
